// Transformer_21955872817210
// MI455X (gfx1250) — compile-verified, matrix operand order checked by reading, not run on hardware
//
#include <hip/hip_runtime.h>
#include <hip/hip_bf16.h>

typedef __attribute__((ext_vector_type(16))) __bf16 v16bf;
typedef __attribute__((ext_vector_type(4)))  __bf16 v4bf;
typedef __attribute__((ext_vector_type(8)))  float  v8f;
typedef __attribute__((ext_vector_type(4)))  unsigned int u32x4;
typedef __attribute__((ext_vector_type(8)))  int i32x8;
typedef __attribute__((ext_vector_type(4)))  int i32x4;

#define BDIM 32
#define SDIM 512
#define EDIM 768
#define HDIM 12
#define HD   64
#define HIDD 3072
#define TOK  (BDIM * SDIM)   // 16384 rows

// ---------------------------------------------------------------------------
// TDM: async DMA of a 2D bf16 tile (tile_d0 elems/row x tile_d1 rows, row
// pitch row_stride elems) from global to LDS. D# packed per CDNA5 ISA §8.3-8.4.
// Completion tracked by TENSORcnt. (6-arg builtin form on this toolchain.)
// ---------------------------------------------------------------------------
__device__ __forceinline__ void tdm_load_2d_bf16(unsigned int lds_addr,
                                                 const void* gaddr,
                                                 unsigned int tile_d0,
                                                 unsigned int tile_d1,
                                                 unsigned int row_stride) {
  const unsigned long long ga = (unsigned long long)(uintptr_t)gaddr;
  u32x4 g0;
  g0[0] = 1u;                                   // count=1, user descriptor
  g0[1] = lds_addr;                             // LDS byte address
  g0[2] = (unsigned int)ga;                     // global_addr[31:0]
  g0[3] = (unsigned int)((ga >> 32) & 0x1FFFFFFu) | (2u << 30);  // [56:32] | type=2
  i32x8 g1;
  g1[0] = (int)(1u << 16);                      // data_size=1 (2B), mask=0, no flags
  g1[1] = (int)((tile_d0 & 0xFFFFu) << 16);     // tensor_dim0[15:0] (== tile)
  g1[2] = (int)(((tile_d0 >> 16) & 0xFFFFu) | ((tile_d1 & 0xFFFFu) << 16));
  g1[3] = (int)(((tile_d1 >> 16) & 0xFFFFu) | ((tile_d0 & 0xFFFFu) << 16)); // tile_dim0
  g1[4] = (int)(tile_d1 & 0xFFFFu);             // tile_dim1 (tile_dim2=0)
  g1[5] = (int)row_stride;                      // tensor_dim0_stride[31:0]
  g1[6] = 0;                                    // stride[47:32]=0, dim1_stride=0
  g1[7] = 0;
  const i32x4 z4 = {0, 0, 0, 0};                // groups 2/3 unused (2D tensor)
  const i32x8 z8 = {0, 0, 0, 0, 0, 0, 0, 0};
  __builtin_amdgcn_tensor_load_to_lds(g0, g1, z4, z4, z8, 0);
}

__device__ __forceinline__ unsigned int lds_addr_of(const void* p) {
  return (unsigned int)(uintptr_t)p;            // LDS aperture: addr[31:0] = LDS offset
}

// ---------------------------------------------------------------------------
// fused fp32 -> bf16 convert + transpose for weights:  in[K][N] -> out[N][K]
// ---------------------------------------------------------------------------
__global__ __launch_bounds__(256)
void cvt_transpose_bf16_kernel(const float* __restrict__ in, __bf16* __restrict__ out,
                               int K, int N) {
  __shared__ float tile[32][33];
  const int k0 = blockIdx.y * 32;
  const int n0 = blockIdx.x * 32;
  const int tx = threadIdx.x & 31;
  const int ty = threadIdx.x >> 5;   // 0..7
  #pragma unroll
  for (int r = ty; r < 32; r += 8)
    tile[r][tx] = in[(size_t)(k0 + r) * N + n0 + tx];
  __syncthreads();
  #pragma unroll
  for (int r = ty; r < 32; r += 8)
    out[(size_t)(n0 + r) * K + k0 + tx] = (__bf16)tile[tx][r];
}

// ---------------------------------------------------------------------------
// LayerNorm over E=768, one block (256 thr) per row, bf16 output for WMMA A
// ---------------------------------------------------------------------------
__global__ __launch_bounds__(256)
void layernorm_bf16_kernel(const float* __restrict__ x, const float* __restrict__ g,
                           const float* __restrict__ be, __bf16* __restrict__ out) {
  const int row = blockIdx.x;
  const int tid = threadIdx.x;
  const float* xr = x + (size_t)row * EDIM;
  float v0 = xr[tid], v1 = xr[tid + 256], v2 = xr[tid + 512];
  __shared__ float rs[256], rq[256];
  rs[tid] = v0 + v1 + v2;
  rq[tid] = v0 * v0 + v1 * v1 + v2 * v2;
  __syncthreads();
  for (int off = 128; off > 0; off >>= 1) {
    if (tid < off) { rs[tid] += rs[tid + off]; rq[tid] += rq[tid + off]; }
    __syncthreads();
  }
  const float mean = rs[0] * (1.0f / EDIM);
  const float var  = rq[0] * (1.0f / EDIM) - mean * mean;
  const float rstd = rsqrtf(var + 1e-5f);
  __bf16* orow = out + (size_t)row * EDIM;
  orow[tid]       = (__bf16)((v0 - mean) * rstd * g[tid]       + be[tid]);
  orow[tid + 256] = (__bf16)((v1 - mean) * rstd * g[tid + 256] + be[tid + 256]);
  orow[tid + 512] = (__bf16)((v2 - mean) * rstd * g[tid + 512] + be[tid + 512]);
}

// ---------------------------------------------------------------------------
// bf16 WMMA GEMM: C[M,N] = A[M,K] * Bt[N,K]^T + bias[N]
// 128x128 block, BK=32. TDM (tensor_load_to_lds) double-buffers the A/B
// 128x32 tiles: wave 0 issues DMA for step s+1 while all 8 waves run WMMA on
// step s, then drains TENSORcnt and the block barriers.
// ---------------------------------------------------------------------------
template <bool OUT_BF16>
__global__ __launch_bounds__(256)
void gemm_bf16_wmma_kernel(const __bf16* __restrict__ A, const __bf16* __restrict__ Bt,
                           const float* __restrict__ bias, void* __restrict__ Cout,
                           int M, int N, int K) {
  __shared__ __bf16 As[2][128 * 32];
  __shared__ __bf16 Bs[2][128 * 32];
  const int tid  = threadIdx.x;
  const int lane = tid & 31;
  const int wave = tid >> 5;
  const int wm = wave >> 1;          // 0..3 -> M offset wm*32
  const int wn = wave & 1;           // 0..1 -> N offset wn*64
  const int m0 = blockIdx.y * 128;
  const int n0 = blockIdx.x * 128;
  const int l15 = lane & 15;
  const int koff = (lane & 16) ? 8 : 0;
  const int rowsel = (lane & 16) ? 8 : 0;

  const __bf16* Ablk = A  + (size_t)m0 * K;
  const __bf16* Bblk = Bt + (size_t)n0 * K;
  unsigned int ldsA[2] = { lds_addr_of(&As[0][0]), lds_addr_of(&As[1][0]) };
  unsigned int ldsB[2] = { lds_addr_of(&Bs[0][0]), lds_addr_of(&Bs[1][0]) };

  v8f acc[2][4];
  #pragma unroll
  for (int i = 0; i < 2; ++i)
    #pragma unroll
    for (int j = 0; j < 4; ++j)
      #pragma unroll
      for (int e = 0; e < 8; ++e) acc[i][j][e] = 0.0f;

  // preload step 0 into buffer 0 via TDM
  if (wave == 0) {
    tdm_load_2d_bf16(ldsA[0], Ablk, 32, 128, (unsigned)K);
    tdm_load_2d_bf16(ldsB[0], Bblk, 32, 128, (unsigned)K);
    __builtin_amdgcn_s_wait_tensorcnt(0);
  }
  __syncthreads();

  const int nsteps = K >> 5;
  for (int s = 0; s < nsteps; ++s) {
    const int cur = s & 1, nxt = cur ^ 1;
    const bool has_next = (s + 1) < nsteps;
    if (has_next && wave == 0) {          // DMA next K-slice during compute
      tdm_load_2d_bf16(ldsA[nxt], Ablk + (size_t)(s + 1) * 32, 32, 128, (unsigned)K);
      tdm_load_2d_bf16(ldsB[nxt], Bblk + (size_t)(s + 1) * 32, 32, 128, (unsigned)K);
    }

    v16bf af[2], bfv[4];
    #pragma unroll
    for (int mt = 0; mt < 2; ++mt) {
      const __bf16* ar = &As[cur][(wm * 32 + mt * 16 + l15) * 32];
      #pragma unroll
      for (int e = 0; e < 8; ++e) {
        af[mt][e]     = ar[koff + e];
        af[mt][e + 8] = ar[16 + koff + e];
      }
    }
    #pragma unroll
    for (int nt = 0; nt < 4; ++nt) {
      const __bf16* br = &Bs[cur][(wn * 64 + nt * 16 + l15) * 32];
      #pragma unroll
      for (int e = 0; e < 8; ++e) {
        bfv[nt][e]     = br[koff + e];
        bfv[nt][e + 8] = br[16 + koff + e];
      }
    }
    #pragma unroll
    for (int mt = 0; mt < 2; ++mt)
      #pragma unroll
      for (int nt = 0; nt < 4; ++nt)
        acc[mt][nt] = __builtin_amdgcn_wmma_f32_16x16x32_bf16(
            false, af[mt], false, bfv[nt], (short)0, acc[mt][nt], false, false);

    if (wave == 0) __builtin_amdgcn_s_wait_tensorcnt(0);
    __syncthreads();
  }

  #pragma unroll
  for (int mt = 0; mt < 2; ++mt) {
    const int rowb = m0 + wm * 32 + mt * 16 + rowsel;
    #pragma unroll
    for (int nt = 0; nt < 4; ++nt) {
      const int col = n0 + wn * 64 + nt * 16 + l15;
      const float bv = bias[col];
      #pragma unroll
      for (int r = 0; r < 8; ++r) {
        const float v = acc[mt][nt][r] + bv;
        const size_t idx = (size_t)(rowb + r) * N + col;
        if (OUT_BF16) ((__bf16*)Cout)[idx] = (__bf16)v;
        else          ((float*)Cout)[idx]  = v;
      }
    }
  }
}

// ---------------------------------------------------------------------------
// Fused attention per (b,h): K,V TDM-staged into LDS (128 KB) + 4 per-wave
// 16x512 f32 score panels (128 KB) = 256 KB of the WGP's 320 KB LDS.
// scores = sqrt(E) * Q K^T ; softmax(softmax(.)) ; O = P V   (all WMMA bf16)
// ---------------------------------------------------------------------------
__global__ __launch_bounds__(128)
void attention_kernel(const __bf16* __restrict__ qkv, __bf16* __restrict__ o) {
  extern __shared__ __align__(16) char smem_raw[];
  __bf16* Ksh = (__bf16*)smem_raw;            // 512*64
  __bf16* Vsh = Ksh + SDIM * HD;              // 512*64
  float*  Pall = (float*)(Vsh + SDIM * HD);   // 4 waves * 16 * 512
  const int tid  = threadIdx.x;
  const int lane = tid & 31;
  const int wave = tid >> 5;
  float* P = Pall + wave * 16 * SDIM;
  const int bh = blockIdx.x;
  const int b  = bh / HDIM;
  const int hh = bh % HDIM;
  const size_t tokbase = (size_t)b * SDIM;
  const int l15 = lane & 15;
  const int koff = (lane & 16) ? 8 : 0;
  const int rowsel = (lane & 16) ? 8 : 0;

  // TDM-stage K and V: 512x64 bf16 tiles, row pitch 2304 elems in qkv
  if (wave == 0) {
    const size_t base = tokbase * (3 * EDIM) + (size_t)hh * (3 * HD);
    tdm_load_2d_bf16(lds_addr_of(Ksh), qkv + base + HD,     HD, SDIM, 3 * EDIM);
    tdm_load_2d_bf16(lds_addr_of(Vsh), qkv + base + 2 * HD, HD, SDIM, 3 * EDIM);
    __builtin_amdgcn_s_wait_tensorcnt(0);
  }
  __syncthreads();

  const float scale = 27.712812921102035f;  // sqrt(768)

  for (int qt = wave; qt < SDIM / 16; qt += 4) {
    const int s0 = qt * 16;
    // Q fragments straight from global (bf16), K-dim = 64 -> two x32 halves
    v16bf qf[2];
    {
      const size_t qbase = (tokbase + s0 + l15) * (3 * EDIM) + (size_t)hh * (3 * HD);
      #pragma unroll
      for (int hseg = 0; hseg < 2; ++hseg) {
        #pragma unroll
        for (int e = 0; e < 8; ++e) {
          qf[hseg][e]     = qkv[qbase + hseg * 32 + koff + e];
          qf[hseg][e + 8] = qkv[qbase + hseg * 32 + 16 + koff + e];
        }
      }
    }
    // scores row-panel [16 x 512]
    for (int nt = 0; nt < SDIM / 16; ++nt) {
      const int t0 = nt * 16;
      v16bf kf[2];
      #pragma unroll
      for (int hseg = 0; hseg < 2; ++hseg) {
        const __bf16* kr = &Ksh[(t0 + l15) * HD + hseg * 32];
        #pragma unroll
        for (int e = 0; e < 8; ++e) {
          kf[hseg][e]     = kr[koff + e];
          kf[hseg][e + 8] = kr[16 + koff + e];
        }
      }
      v8f acc;
      #pragma unroll
      for (int e = 0; e < 8; ++e) acc[e] = 0.0f;
      acc = __builtin_amdgcn_wmma_f32_16x16x32_bf16(false, qf[0], false, kf[0],
                                                    (short)0, acc, false, false);
      acc = __builtin_amdgcn_wmma_f32_16x16x32_bf16(false, qf[1], false, kf[1],
                                                    (short)0, acc, false, false);
      #pragma unroll
      for (int r = 0; r < 8; ++r)
        P[(rowsel + r) * SDIM + t0 + l15] = acc[r] * scale;
    }
    __syncthreads();

    // double softmax: 2 lanes per row, 256 cols (64 float4) each,
    // pair-combined via cross-lane shfl
    {
      float4* pv = (float4*)(P + l15 * SDIM + (lane >> 4) * 256);
      #pragma unroll
      for (int pass = 0; pass < 2; ++pass) {
        float mx = -3.4e38f;
        for (int j = 0; j < 64; ++j) {
          const float4 t = pv[j];
          mx = fmaxf(mx, fmaxf(fmaxf(t.x, t.y), fmaxf(t.z, t.w)));
        }
        mx = fmaxf(mx, __shfl_xor(mx, 16, 32));
        float sum = 0.0f;
        for (int j = 0; j < 64; ++j) {
          float4 t = pv[j];
          t.x = __expf(t.x - mx); t.y = __expf(t.y - mx);
          t.z = __expf(t.z - mx); t.w = __expf(t.w - mx);
          pv[j] = t;
          sum += (t.x + t.y) + (t.z + t.w);
        }
        sum += __shfl_xor(sum, 16, 32);
        const float inv = 1.0f / sum;
        for (int j = 0; j < 64; ++j) {
          float4 t = pv[j];
          t.x *= inv; t.y *= inv; t.z *= inv; t.w *= inv;
          pv[j] = t;
        }
        __syncthreads();
      }
    }

    // O = P[16x512] @ V[512x64]
    v8f oacc[4];
    #pragma unroll
    for (int d0 = 0; d0 < 4; ++d0)
      #pragma unroll
      for (int e = 0; e < 8; ++e) oacc[d0][e] = 0.0f;
    for (int kt = 0; kt < 16; ++kt) {
      const int tb = kt * 32;
      v16bf pf;
      {
        const float* pr = P + l15 * SDIM + tb;
        #pragma unroll
        for (int e = 0; e < 8; ++e) {
          pf[e]     = (__bf16)pr[koff + e];
          pf[e + 8] = (__bf16)pr[16 + koff + e];
        }
      }
      #pragma unroll
      for (int d0 = 0; d0 < 4; ++d0) {
        v16bf vf;
        #pragma unroll
        for (int e = 0; e < 8; ++e) {
          vf[e]     = Vsh[(tb + koff + e) * HD + d0 * 16 + l15];
          vf[e + 8] = Vsh[(tb + 16 + koff + e) * HD + d0 * 16 + l15];
        }
        oacc[d0] = __builtin_amdgcn_wmma_f32_16x16x32_bf16(
            false, pf, false, vf, (short)0, oacc[d0], false, false);
      }
    }
    #pragma unroll
    for (int d0 = 0; d0 < 4; ++d0) {
      const int d = d0 * 16 + l15;
      #pragma unroll
      for (int r = 0; r < 8; ++r)
        o[(tokbase + s0 + rowsel + r) * EDIM + hh * HD + d] = (__bf16)oacc[d0][r];
    }
    __syncthreads();
  }
}

// ---------------------------------------------------------------------------
// elementwise helpers (128-bit float4 / 64-bit 4 x bf16 accesses)
// ---------------------------------------------------------------------------
__global__ void add_inplace_kernel(float* __restrict__ dst,
                                   const float* __restrict__ src, size_t n4) {
  size_t i = (size_t)blockIdx.x * blockDim.x + threadIdx.x;
  const size_t stride = (size_t)gridDim.x * blockDim.x;
  float4* d = (float4*)dst;
  const float4* s = (const float4*)src;
  for (; i < n4; i += stride) {
    float4 a = d[i];
    const float4 b = s[i];
    a.x += b.x; a.y += b.y; a.z += b.z; a.w += b.w;
    d[i] = a;
  }
}

__global__ void reglu_kernel(const __bf16* __restrict__ h0,
                             __bf16* __restrict__ hm, size_t n4) {  // n4 = TOK*HIDD/4
  size_t i = (size_t)blockIdx.x * blockDim.x + threadIdx.x;
  const size_t stride = (size_t)gridDim.x * blockDim.x;
  for (; i < n4; i += stride) {
    const size_t r = i / (HIDD / 4), j4 = i % (HIDD / 4);
    const v4bf a4 = *(const v4bf*)&h0[r * (2 * HIDD) + j4 * 4];
    const v4bf b4 = *(const v4bf*)&h0[r * (2 * HIDD) + HIDD + j4 * 4];
    v4bf o4;
    #pragma unroll
    for (int e = 0; e < 4; ++e)
      o4[e] = (__bf16)((float)a4[e] * fmaxf((float)b4[e], 0.0f));
    *(v4bf*)&hm[i * 4] = o4;
  }
}

// ---------------------------------------------------------------------------
// launch
// ---------------------------------------------------------------------------
extern "C" void kernel_launch(void* const* d_in, const int* in_sizes, int n_in,
                              void* d_out, int out_size, void* d_ws, size_t ws_size,
                              hipStream_t stream) {
  const float* x     = (const float*)d_in[0];
  const float* W_qkv = (const float*)d_in[1];
  const float* b_qkv = (const float*)d_in[2];
  const float* W_out = (const float*)d_in[3];
  const float* b_out = (const float*)d_in[4];
  const float* W0    = (const float*)d_in[5];
  const float* b0    = (const float*)d_in[6];
  const float* W1    = (const float*)d_in[7];
  const float* b1    = (const float*)d_in[8];
  const float* g0    = (const float*)d_in[9];
  const float* be0   = (const float*)d_in[10];
  const float* g1    = (const float*)d_in[11];
  const float* be1   = (const float*)d_in[12];
  (void)in_sizes; (void)n_in; (void)out_size; (void)ws_size;

  char* wsp = (char*)d_ws;
  auto alloc = [&](size_t bytes) {
    char* p = wsp;
    wsp += (bytes + 255) & ~(size_t)255;
    return p;
  };
  __bf16* wqkv_bt = (__bf16*)alloc((size_t)EDIM * 3 * EDIM * 2);   // [3E][E]
  __bf16* wout_bt = (__bf16*)alloc((size_t)EDIM * EDIM * 2);       // [E][E]
  __bf16* w0_bt   = (__bf16*)alloc((size_t)EDIM * 2 * HIDD * 2);   // [2H][E]
  __bf16* w1_bt   = (__bf16*)alloc((size_t)HIDD * EDIM * 2);       // [E][H]
  __bf16* h_bf    = (__bf16*)alloc((size_t)TOK * EDIM * 2);
  __bf16* qkv_bf  = (__bf16*)alloc((size_t)TOK * 3 * EDIM * 2);
  __bf16* o_bf    = (__bf16*)alloc((size_t)TOK * EDIM * 2);
  float*  x1_f    = (float*)alloc((size_t)TOK * EDIM * 4);
  __bf16* h0_bf   = (__bf16*)alloc((size_t)TOK * 2 * HIDD * 2);
  __bf16* hmid_bf = (__bf16*)alloc((size_t)TOK * HIDD * 2);
  float*  out_f   = (float*)d_out;

  // weight convert + transpose (fp32 [K][N] -> bf16 [N][K])
  cvt_transpose_bf16_kernel<<<dim3(3 * EDIM / 32, EDIM / 32), 256, 0, stream>>>(
      W_qkv, wqkv_bt, EDIM, 3 * EDIM);
  cvt_transpose_bf16_kernel<<<dim3(EDIM / 32, EDIM / 32), 256, 0, stream>>>(
      W_out, wout_bt, EDIM, EDIM);
  cvt_transpose_bf16_kernel<<<dim3(2 * HIDD / 32, EDIM / 32), 256, 0, stream>>>(
      W0, w0_bt, EDIM, 2 * HIDD);
  cvt_transpose_bf16_kernel<<<dim3(EDIM / 32, HIDD / 32), 256, 0, stream>>>(
      W1, w1_bt, HIDD, EDIM);

  // LN0 -> h_bf
  layernorm_bf16_kernel<<<TOK, 256, 0, stream>>>(x, g0, be0, h_bf);

  // QKV GEMM: [16384,768] x [768,2304] -> bf16
  gemm_bf16_wmma_kernel<true><<<dim3(3 * EDIM / 128, TOK / 128), 256, 0, stream>>>(
      h_bf, wqkv_bt, b_qkv, (void*)qkv_bf, TOK, 3 * EDIM, EDIM);

  // fused attention (256 KB dynamic LDS per workgroup)
  {
    const size_t att_lds = (size_t)2 * SDIM * HD * 2 + (size_t)4 * 16 * SDIM * 4;
    attention_kernel<<<BDIM * HDIM, 128, att_lds, stream>>>(qkv_bf, o_bf);
  }

  // out projection: [16384,768] x [768,768] -> f32 (into x1)
  gemm_bf16_wmma_kernel<false><<<dim3(EDIM / 128, TOK / 128), 256, 0, stream>>>(
      o_bf, wout_bt, b_out, (void*)x1_f, TOK, EDIM, EDIM);

  // x1 = x + attn_out
  add_inplace_kernel<<<2048, 256, 0, stream>>>(x1_f, x, (size_t)TOK * EDIM / 4);

  // LN1 -> h_bf (reuse)
  layernorm_bf16_kernel<<<TOK, 256, 0, stream>>>(x1_f, g1, be1, h_bf);

  // FFN W0: [16384,768] x [768,6144] -> bf16
  gemm_bf16_wmma_kernel<true><<<dim3(2 * HIDD / 128, TOK / 128), 256, 0, stream>>>(
      h_bf, w0_bt, b0, (void*)h0_bf, TOK, 2 * HIDD, EDIM);

  // ReGLU -> bf16 [16384,3072]
  reglu_kernel<<<4096, 256, 0, stream>>>(h0_bf, hmid_bf, (size_t)TOK * HIDD / 4);

  // FFN W1: [16384,3072] x [3072,768] -> f32 into d_out
  gemm_bf16_wmma_kernel<false><<<dim3(EDIM / 128, TOK / 128), 256, 0, stream>>>(
      hmid_bf, w1_bt, b1, (void*)out_f, TOK, EDIM, HIDD);

  // out += x1 (final residual)
  add_inplace_kernel<<<2048, 256, 0, stream>>>(out_f, x1_f, (size_t)TOK * EDIM / 4);
}
